// GPRGNNAugmented_11209864643036
// MI455X (gfx1250) — compile-verified, matrix operand order checked by reading, not run on hardware
//
#include <hip/hip_runtime.h>
#include <hip/hip_bf16.h>

typedef __attribute__((ext_vector_type(16))) __bf16 v16bf;
typedef __attribute__((ext_vector_type(8)))  float  v8f;

// ---------------------------------------------------------------------------
// Kernel 0: one-time pack of W1[256,512] and W2[64,256] f32 -> bf16 in ws.
// ---------------------------------------------------------------------------
__global__ __launch_bounds__(256)
void pack_weights(const float* __restrict__ w1, const float* __restrict__ w2,
                  __bf16* __restrict__ w1b, __bf16* __restrict__ w2b,
                  int n1, int n2)
{
    const int i = blockIdx.x * 256 + threadIdx.x;
    if (i < n1) w1b[i] = (__bf16)w1[i];
    else if (i < n1 + n2) w2b[i - n1] = (__bf16)w2[i - n1];
}

// ---------------------------------------------------------------------------
// Kernel 1: fused MLP  out[N,64] = relu(feat @ W1^T + b1) @ W2^T + b2
// One block = 16 rows, 256 threads = 8 wave32.
// GEMM1: wave w computes column tiles [32w,32w+16) and [32w+16,32w+32).
// GEMM2: waves 0..3 compute the 4 column tiles of the 64-wide output.
// B fragments load bf16 weights directly (no converts in the hot loop).
// ---------------------------------------------------------------------------
__global__ __launch_bounds__(256)
void mlp_wmma(const float* __restrict__ feat, const __bf16* __restrict__ w1b,
              const float* __restrict__ b1v, const __bf16* __restrict__ w2b,
              const float* __restrict__ b2v, float* __restrict__ out, int nrows)
{
    __shared__ __bf16 sA[16 * 512];   // 16 KB: 16 feature rows, bf16
    __shared__ __bf16 sH[16 * 256];   // 8 KB:  hidden after relu, bf16

    const int tid     = threadIdx.x;
    const int rowBase = blockIdx.x * 16;

    // cooperative load of the 16-row strip (contiguous 8192 floats), f32->bf16
    const float* fp = feat + (size_t)rowBase * 512;
    for (int i = tid; i < 16 * 512; i += 256) sA[i] = (__bf16)fp[i];
    __syncthreads();

    const int wave = tid >> 5;
    const int l    = tid & 31;
    const int m    = l & 15;          // M (A) or N (B/C/D) index for this lane
    const int kb   = (l >> 4) * 8;    // K base for A/B fragments
    const int mofs = (l >> 4) * 8;    // M offset for C/D rows

    // ---------------- GEMM1: [16,512] x [512,256] ----------------
    v8f acc0 = {}; v8f acc1 = {};
    const int c0 = wave * 32 + m;
    const int c1 = wave * 32 + 16 + m;
    const __bf16* w1r0 = w1b + (size_t)c0 * 512;
    const __bf16* w1r1 = w1b + (size_t)c1 * 512;
    const __bf16* arow = sA + m * 512;

    #pragma unroll
    for (int k0 = 0; k0 < 512; k0 += 32) {
        v16bf a, bb0, bb1;
        #pragma unroll
        for (int e = 0; e < 8; ++e) {
            a[e]       = arow[k0 + kb + e];
            a[e + 8]   = arow[k0 + kb + 16 + e];
            bb0[e]     = w1r0[k0 + kb + e];
            bb0[e + 8] = w1r0[k0 + kb + 16 + e];
            bb1[e]     = w1r1[k0 + kb + e];
            bb1[e + 8] = w1r1[k0 + kb + 16 + e];
        }
        acc0 = __builtin_amdgcn_wmma_f32_16x16x32_bf16(false, a, false, bb0,
                                                       (short)0, acc0, false, false);
        acc1 = __builtin_amdgcn_wmma_f32_16x16x32_bf16(false, a, false, bb1,
                                                       (short)0, acc1, false, false);
    }

    const float bias0 = b1v[c0];
    const float bias1 = b1v[c1];
    #pragma unroll
    for (int r = 0; r < 8; ++r) {
        const int M = r + mofs;
        float v0 = acc0[r] + bias0; v0 = v0 > 0.f ? v0 : 0.f;
        float v1 = acc1[r] + bias1; v1 = v1 > 0.f ? v1 : 0.f;
        sH[M * 256 + c0] = (__bf16)v0;
        sH[M * 256 + c1] = (__bf16)v1;
    }
    __syncthreads();

    // ---------------- GEMM2: [16,256] x [256,64] ----------------
    if (wave < 4) {
        v8f acc = {};
        const int cc = wave * 16 + m;
        const __bf16* w2r  = w2b + (size_t)cc * 256;
        const __bf16* hrow = sH + m * 256;
        #pragma unroll
        for (int k0 = 0; k0 < 256; k0 += 32) {
            v16bf a, bb;
            #pragma unroll
            for (int e = 0; e < 8; ++e) {
                a[e]      = hrow[k0 + kb + e];
                a[e + 8]  = hrow[k0 + kb + 16 + e];
                bb[e]     = w2r[k0 + kb + e];
                bb[e + 8] = w2r[k0 + kb + 16 + e];
            }
            acc = __builtin_amdgcn_wmma_f32_16x16x32_bf16(false, a, false, bb,
                                                          (short)0, acc, false, false);
        }
        const float bias = b2v[cc];
        #pragma unroll
        for (int r = 0; r < 8; ++r) {
            const int row = rowBase + r + mofs;
            if (row < nrows) out[(size_t)row * 64 + cc] = acc[r] + bias;
        }
    }
}

// ---------------------------------------------------------------------------
// Kernel 2: edge scatter  nxt[dst] += norm * cur[src]   (C = 64 floats/row)
// 16 lanes per edge, each lane handles 4 columns (float4 gather + 4 atomics).
// Edge index/norm streams are marked non-temporal so the 192MB L2 stays
// dedicated to the node matrices (gather + atomic working set).
// ---------------------------------------------------------------------------
__global__ __launch_bounds__(256)
void scatter_edges(const int* __restrict__ src, const int* __restrict__ dst,
                   const float* __restrict__ nrm, const float* __restrict__ cur,
                   float* __restrict__ nxt, int E)
{
    const int t   = blockIdx.x * 256 + threadIdx.x;
    const int e   = t >> 4;
    const int sub = t & 15;
    if (e >= E) return;

    const long  s = __builtin_nontemporal_load(src + e);
    const long  d = __builtin_nontemporal_load(dst + e);
    const float w = __builtin_nontemporal_load(nrm + e);

    const float4 v = *(const float4*)(cur + s * 64 + sub * 4);
    float* o = nxt + d * 64 + sub * 4;
    unsafeAtomicAdd(o + 0, w * v.x);
    unsafeAtomicAdd(o + 1, w * v.y);
    unsafeAtomicAdd(o + 2, w * v.z);
    unsafeAtomicAdd(o + 3, w * v.w);
}

// ---------------------------------------------------------------------------
// Kernel 3: init  out = (t1[0]+t2[0])*x ; cur2 = x ; tmp1 = tmp2 = 0
// ---------------------------------------------------------------------------
__global__ __launch_bounds__(256)
void init_prop(const float* __restrict__ x, float* __restrict__ cur2,
               float* __restrict__ out, float* __restrict__ tmp1,
               float* __restrict__ tmp2, const float* __restrict__ t1,
               const float* __restrict__ t2, int n)
{
    const int i = blockIdx.x * 256 + threadIdx.x;
    if (i >= n) return;
    const float s = t1[0] + t2[0];
    const float xv = x[i];
    out[i]  = s * xv;
    cur2[i] = xv;
    tmp1[i] = 0.f;
    tmp2[i] = 0.f;
}

// ---------------------------------------------------------------------------
// Kernel 4: out += temp[k]*srcv ; zbuf = 0   (zbuf becomes next hop's target)
// ---------------------------------------------------------------------------
__global__ __launch_bounds__(256)
void axpy_zero(float* __restrict__ out, const float* __restrict__ srcv,
               float* __restrict__ zbuf, const float* __restrict__ temp,
               int kidx, int n)
{
    const int i = blockIdx.x * 256 + threadIdx.x;
    if (i >= n) return;
    const float t = temp[kidx];
    out[i] += t * srcv[i];
    zbuf[i] = 0.f;
}

extern "C" void kernel_launch(void* const* d_in, const int* in_sizes, int n_in,
                              void* d_out, int out_size, void* d_ws, size_t ws_size,
                              hipStream_t stream) {
    const float* feature = (const float*)d_in[0];
    const int*   ei1     = (const int*)  d_in[1];
    const float* normA   = (const float*)d_in[2];
    const int*   ei2     = (const int*)  d_in[3];
    const float* normA2  = (const float*)d_in[4];
    const float* W1      = (const float*)d_in[5];
    const float* b1v     = (const float*)d_in[6];
    const float* W2      = (const float*)d_in[7];
    const float* b2v     = (const float*)d_in[8];
    const float* temp1   = (const float*)d_in[9];
    const float* temp2   = (const float*)d_in[10];

    const int E  = in_sizes[2];           // 3,200,000
    const int N  = in_sizes[0] / 512;     // 100,000
    const int K  = in_sizes[9] - 1;       // 10
    const int NC = N * 64;
    const int n1 = in_sizes[5];           // 256*512
    const int n2 = in_sizes[7];           // 64*256

    float* out = (float*)d_out;

    // workspace: 4 node buffers of N*64 f32 (25.6 MB each) + bf16 weights
    float* cur1 = (float*)d_ws;
    float* tmp1 = cur1 + NC;
    float* cur2 = tmp1 + NC;
    float* tmp2 = cur2 + NC;
    __bf16* w1b = (__bf16*)(tmp2 + NC);
    __bf16* w2b = w1b + n1;

    const int* src1 = ei1;  const int* dst1 = ei1 + E;
    const int* src2 = ei2;  const int* dst2 = ei2 + E;

    // 0) pack weights to bf16 (tiny, once per launch)
    pack_weights<<<(n1 + n2 + 255) / 256, 256, 0, stream>>>(W1, W2, w1b, w2b, n1, n2);

    // 1) fused MLP -> cur1
    mlp_wmma<<<(N + 15) / 16, 256, 0, stream>>>(feature, w1b, b1v, w2b, b2v, cur1, N);

    // 2) out = (t1[0]+t2[0])*x ; cur2 = x ; tmp buffers zeroed
    init_prop<<<(NC + 255) / 256, 256, 0, stream>>>(cur1, cur2, out, tmp1, tmp2,
                                                    temp1, temp2, NC);

    // 3) K hops of each propagation chain
    const int sgrid = (E * 16 + 255) / 256;
    const int egrid = (NC + 255) / 256;
    for (int k = 0; k < K; ++k) {
        scatter_edges<<<sgrid, 256, 0, stream>>>(src1, dst1, normA, cur1, tmp1, E);
        axpy_zero<<<egrid, 256, 0, stream>>>(out, tmp1, cur1, temp1, k + 1, NC);
        { float* t = cur1; cur1 = tmp1; tmp1 = t; }

        scatter_edges<<<sgrid, 256, 0, stream>>>(src2, dst2, normA2, cur2, tmp2, E);
        axpy_zero<<<egrid, 256, 0, stream>>>(out, tmp2, cur2, temp2, k + 1, NC);
        { float* t = cur2; cur2 = tmp2; tmp2 = t; }
    }
    (void)n_in; (void)out_size; (void)ws_size;
}
